// PlanNetwork_15049565405659
// MI455X (gfx1250) — matrix operand order repaired and checked by reading, not compiled, not run on hardware
//
#include <hip/hip_runtime.h>
#include <cstdint>
#include <cstddef>

// Problem constants (match reference)
#define HD   512
#define FD   2048
#define BB   32
#define NN   256
#define NHE  8
#define DD   64
#define LL   8
#define MROWS (BB * NN)   // 8192

typedef __bf16 v16bf __attribute__((ext_vector_type(16)));
typedef float  v8f   __attribute__((ext_vector_type(8)));
typedef int    v4i   __attribute__((ext_vector_type(4)));
typedef unsigned short u16;
typedef unsigned int   u32;

struct B32x2 { uint4 a, b; };

__device__ __forceinline__ u16 f2bf(float f) {
  u32 u = __builtin_bit_cast(u32, f);
  u32 r = u + 0x7FFFu + ((u >> 16) & 1u);   // round-to-nearest-even
  return (u16)(r >> 16);
}

__device__ __forceinline__ v16bf make_frag(uint4 a, uint4 b) {
  B32x2 t{a, b};
  return __builtin_bit_cast(v16bf, t);
}

__device__ __forceinline__ v8f wmma_bf16(v16bf a, v16bf b, v8f c) {
  // D = A x B + C ; 16x16x32 bf16 -> f32, wave32
  return __builtin_amdgcn_wmma_f32_16x16x32_bf16(
      false, a, false, b, (short)0, c, false, false);
}

// CDNA5 async global->LDS path (ASYNCcnt), guarded so absence cannot break build
#if defined(__AMDGCN__) && \
    __has_builtin(__builtin_amdgcn_global_load_async_to_lds_b128) && \
    __has_builtin(__builtin_amdgcn_s_wait_asynccnt)
#define USE_ASYNC_LDS 1
#else
#define USE_ASYNC_LDS 0
#endif

#if USE_ASYNC_LDS
__device__ __forceinline__ void async_b128(const void* g, void* l) {
  // builtin takes v4i* pointers (generic); backend infers AS from provenance
  __builtin_amdgcn_global_load_async_to_lds_b128((v4i*)g, (v4i*)l, 0, 0);
}
#endif

// ---------------------------------------------------------------------------
// Weight conversion: dst[n*K + k] = bf16(src[k*Nout + n] * scale)   (transpose)
// ---------------------------------------------------------------------------
__global__ void conv_w(const float* __restrict__ src, u16* __restrict__ dst,
                       int K, int Nout, float scale) {
  int i = blockIdx.x * 256 + threadIdx.x;
  int total = K * Nout;
  if (i >= total) return;
  int n = i / K, kk = i - n * K;
  dst[i] = f2bf(src[(size_t)kk * Nout + n] * scale);
}

__global__ void scale_copy(const float* __restrict__ src, float* __restrict__ dst,
                           int n, float scale) {
  int i = blockIdx.x * 256 + threadIdx.x;
  if (i < n) dst[i] = src[i] * scale;
}

// ---------------------------------------------------------------------------
// LayerNorm: one wave (32 lanes) per row of 512, output bf16
// ---------------------------------------------------------------------------
__global__ __launch_bounds__(256)
void ln_kernel(const float* __restrict__ x, const float* __restrict__ sc,
               const float* __restrict__ bi, u16* __restrict__ y) {
  const int lane = threadIdx.x & 31, wid = threadIdx.x >> 5;
  const int row  = blockIdx.x * 8 + wid;
  const float* xr = x + (size_t)row * HD;
  float v[16];
  float s = 0.0f;
#pragma unroll
  for (int i = 0; i < 16; ++i) { v[i] = xr[i * 32 + lane]; s += v[i]; }
#pragma unroll
  for (int m = 1; m < 32; m <<= 1) s += __shfl_xor(s, m, 32);
  const float mu = s * (1.0f / 512.0f);
  float var = 0.0f;
#pragma unroll
  for (int i = 0; i < 16; ++i) { float d = v[i] - mu; var += d * d; }
#pragma unroll
  for (int m = 1; m < 32; m <<= 1) var += __shfl_xor(var, m, 32);
  const float inv = rsqrtf(var * (1.0f / 512.0f) + 1e-5f);
  u16* yr = y + (size_t)row * HD;
#pragma unroll
  for (int i = 0; i < 16; ++i) {
    int c = i * 32 + lane;
    yr[c] = f2bf((v[i] - mu) * inv * sc[c] + bi[c]);
  }
}

// ---------------------------------------------------------------------------
// Generic WMMA GEMM: C[M,N] = A[M,K](bf16) * WT[N,K](bf16)^T + bias
// Block tile 128x128, 8 waves (4x2), wave tile 32x64 (2x4 WMMA accumulators).
// k-step 64; XOR-swizzled LDS (64 KB, A+B double buffered); async global->LDS
// pipeline on CDNA5 (ASYNCcnt), register-prefetch single buffer otherwise.
// Modes: 0 = QKV scatter (q,k row-major per head, v transposed)
//        1 = residual(f32) + bias -> f32 out (in-place residual safe)
//        2 = bias + exact GELU -> bf16 out
// ---------------------------------------------------------------------------
#define GEMM_QKV    0
#define GEMM_RESF32 1
#define GEMM_GELU   2

__global__ __launch_bounds__(256)
void gemm_wmma(const u16* __restrict__ A, const u16* __restrict__ WT,
               const float* __restrict__ bias, const float* __restrict__ residual,
               float* __restrict__ outF, u16* __restrict__ outB,
               u16* __restrict__ outQ, u16* __restrict__ outK, u16* __restrict__ outV,
               int M, int N, int K, int mode) {
  // [buf][ A:128x64 | B:128x64 ] bf16, XOR-swizzled rows, 65536 bytes total
  __shared__ __align__(16) u16 sm[2 * 16384];

  const int tid  = threadIdx.x;
  const int lane = tid & 31, wid = tid >> 5;
  const int half = lane >> 4, l16 = lane & 15;
  const int waveM = wid >> 1, waveN = wid & 1;
  const int m0 = blockIdx.y * 128, n0 = blockIdx.x * 128;

  v8f acc[2][4];
#pragma unroll
  for (int mt = 0; mt < 2; ++mt)
#pragma unroll
    for (int nt = 0; nt < 4; ++nt)
      acc[mt][nt] = (v8f){0.f,0.f,0.f,0.f,0.f,0.f,0.f,0.f};

  // swizzled element index within a 128x64 tile (granule = 8 ushorts = 16B)
  auto sidx = [](int row, int o) { return row * 64 + (o ^ ((row & 7) << 3)); };

  // staging map: 256 threads x (4 rows x 16B) per tile
  const int rr  = tid >> 3;          // 0..31
  const int off = (tid & 7) * 8;     // element offset in 64-wide k-slice

  const int nk = K >> 6;

  auto compute = [&](int buf) {
    const u16* sA = sm + buf * 16384;
    const u16* sB = sA + 8192;
#pragma unroll
    for (int kc = 0; kc < 2; ++kc) {
      v16bf af[2], bfr[4];
#pragma unroll
      for (int mt = 0; mt < 2; ++mt) {
        int r = waveM * 32 + mt * 16 + l16;
        af[mt] = make_frag(*(const uint4*)&sA[sidx(r, kc * 32 + half * 8)],
                           *(const uint4*)&sA[sidx(r, kc * 32 + 16 + half * 8)]);
      }
#pragma unroll
      for (int nt = 0; nt < 4; ++nt) {
        int r = waveN * 64 + nt * 16 + l16;
        bfr[nt] = make_frag(*(const uint4*)&sB[sidx(r, kc * 32 + half * 8)],
                            *(const uint4*)&sB[sidx(r, kc * 32 + 16 + half * 8)]);
      }
#pragma unroll
      for (int mt = 0; mt < 2; ++mt)
#pragma unroll
        for (int nt = 0; nt < 4; ++nt)
          acc[mt][nt] = wmma_bf16(af[mt], bfr[nt], acc[mt][nt]);
    }
  };

#if USE_ASYNC_LDS
  // ---- double-buffered async pipeline: 8 async b128 per k-slice per thread
  auto stage = [&](int kk, int buf) {
    u16* dA = sm + buf * 16384;
    u16* dB = dA + 8192;
#pragma unroll
    for (int i = 0; i < 4; ++i) {
      int row = rr + i * 32;
      async_b128(A  + (size_t)(m0 + row) * K + kk + off, dA + sidx(row, off));
      async_b128(WT + (size_t)(n0 + row) * K + kk + off, dB + sidx(row, off));
    }
  };
  stage(0, 0);
  for (int i = 0; i < nk; ++i) {
    const int buf = i & 1;
    if (i + 1 < nk) {
      stage((i + 1) << 6, buf ^ 1);
      __builtin_amdgcn_s_wait_asynccnt(8);   // in-order => slice i resident
    } else {
      __builtin_amdgcn_s_wait_asynccnt(0);
    }
    __syncthreads();                          // everyone's slice i visible
    compute(buf);
    __syncthreads();                          // buf free for slice i+2
  }
#else
  // ---- fallback: register prefetch + single LDS buffer
  uint4 ra[4], rb[4];
  auto gload = [&](int kk) {
#pragma unroll
    for (int i = 0; i < 4; ++i) {
      int row = rr + i * 32;
      ra[i] = *(const uint4*)(A  + (size_t)(m0 + row) * K + kk + off);
      rb[i] = *(const uint4*)(WT + (size_t)(n0 + row) * K + kk + off);
    }
  };
  gload(0);
  for (int i = 0; i < nk; ++i) {
    __syncthreads();
    u16* dA = sm;
    u16* dB = sm + 8192;
#pragma unroll
    for (int j = 0; j < 4; ++j) {
      int row = rr + j * 32;
      *(uint4*)&dA[sidx(row, off)] = ra[j];
      *(uint4*)&dB[sidx(row, off)] = rb[j];
    }
    if (i + 1 < nk) gload((i + 1) << 6);
    __syncthreads();
    compute(0);
  }
#endif

  // Epilogue. C/D layout: lane -> col = lane%16 ; VGPR j -> row = j + 8*(lane/16)
#pragma unroll
  for (int mt = 0; mt < 2; ++mt) {
#pragma unroll
    for (int nt = 0; nt < 4; ++nt) {
      const int colg = n0 + waveN * 64 + nt * 16 + l16;
      const float bv = bias[colg];
#pragma unroll
      for (int j = 0; j < 8; ++j) {
        const int rowg = m0 + waveM * 32 + mt * 16 + j + 8 * half;
        float v = acc[mt][nt][j] + bv;
        if (mode == GEMM_RESF32) {
          const size_t o = (size_t)rowg * N + colg;
          outF[o] = residual[o] + v;
        } else if (mode == GEMM_GELU) {
          float g = 0.5f * v * (1.0f + erff(v * 0.70710678f));
          outB[(size_t)rowg * N + colg] = f2bf(g);
        } else { // GEMM_QKV scatter
          const int b = rowg >> 8, r = rowg & 255;
          const int which = colg >> 9, h = (colg >> 6) & 7, d = colg & 63;
          const int bh = b * 8 + h;
          const u16 pv = f2bf(v);
          if (which == 0)      outQ[((size_t)bh * 256 + r) * 64 + d] = pv;
          else if (which == 1) outK[((size_t)bh * 256 + r) * 64 + d] = pv;
          else                 outV[((size_t)bh * 64 + d) * 256 + r] = pv; // v^T
        }
      }
    }
  }
}

// ---------------------------------------------------------------------------
// Attention: block = (b,h, q-tile of 128); 8 waves x 16 query rows each.
// scores(16x256) via WMMA with bias as C; softmax via half-wave shfl; P->LDS
// (bf16, XOR-swizzled) re-read as A-fragments; ctx = P x v via WMMA (vT input).
// ---------------------------------------------------------------------------
__global__ __launch_bounds__(256)
void attn_wmma(const u16* __restrict__ q, const u16* __restrict__ k,
               const u16* __restrict__ vT, const float* __restrict__ bias,
               u16* __restrict__ ctx) {
  __shared__ u16 Pbuf[8 * 16 * 256];   // 64 KB

  const int tid  = threadIdx.x;
  const int lane = tid & 31, wid = tid >> 5;
  const int half = lane >> 4, l16 = lane & 15;
  const int qt = blockIdx.x & 1, bh = blockIdx.x >> 1;
  const int rowbase = qt * 128 + wid * 16;     // wave's first query row

  const u16*   qbase = q    + (size_t)bh * 256 * 64;
  const u16*   kbase = k    + (size_t)bh * 256 * 64;
  const u16*   vbase = vT   + (size_t)bh * 64 * 256;
  const float* bbase = bias + (size_t)bh * 256 * 256;

  // q A-fragments (D=64 -> 2 k-chunks of 32)
  v16bf qa[2];
  {
    const u16* qp = qbase + (size_t)(rowbase + l16) * 64;
#pragma unroll
    for (int kc = 0; kc < 2; ++kc)
      qa[kc] = make_frag(*(const uint4*)(qp + kc * 32 + half * 8),
                         *(const uint4*)(qp + kc * 32 + 16 + half * 8));
  }

  // scores init = attn_bias (C operand), then += q k^T
  v8f s[16];
#pragma unroll
  for (int t = 0; t < 16; ++t)
#pragma unroll
    for (int j = 0; j < 8; ++j)
      s[t][j] = bbase[(size_t)(rowbase + j + 8 * half) * 256 + t * 16 + l16];

#pragma unroll
  for (int t = 0; t < 16; ++t) {
    const u16* kp = kbase + (size_t)(t * 16 + l16) * 64;
    v16bf kb0 = make_frag(*(const uint4*)(kp + half * 8),
                          *(const uint4*)(kp + 16 + half * 8));
    v16bf kb1 = make_frag(*(const uint4*)(kp + 32 + half * 8),
                          *(const uint4*)(kp + 32 + 16 + half * 8));
    s[t] = wmma_bf16(qa[0], kb0, s[t]);
    s[t] = wmma_bf16(qa[1], kb1, s[t]);
  }

  // row softmax: row data for (j,half) lives in 16 lanes of one half-wave
  u16* pw = Pbuf + wid * (16 * 256);
#pragma unroll
  for (int j = 0; j < 8; ++j) {
    float m = -3.0e38f;
#pragma unroll
    for (int t = 0; t < 16; ++t) m = fmaxf(m, s[t][j]);
#pragma unroll
    for (int msk = 1; msk < 16; msk <<= 1) m = fmaxf(m, __shfl_xor(m, msk, 32));
    float sum = 0.0f;
#pragma unroll
    for (int t = 0; t < 16; ++t) { float e = __expf(s[t][j] - m); s[t][j] = e; sum += e; }
#pragma unroll
    for (int msk = 1; msk < 16; msk <<= 1) sum += __shfl_xor(sum, msk, 32);
    const float inv = 1.0f / sum;
    const int row = j + 8 * half;
    const int sw  = (row & 7) << 3;      // XOR swizzle (8-ushort granules)
#pragma unroll
    for (int t = 0; t < 16; ++t) {
      int col = t * 16 + l16;
      pw[row * 256 + (col ^ sw)] = f2bf(s[t][j] * inv);
    }
  }
  __syncthreads();

  // ctx(16x64) = P(16x256) x v(256x64); B-fragments contiguous from vT
  v8f c[4];
#pragma unroll
  for (int nt = 0; nt < 4; ++nt) c[nt] = (v8f){0.f,0.f,0.f,0.f,0.f,0.f,0.f,0.f};

  const int arow = l16;
  const int asw  = (arow & 7) << 3;
#pragma unroll
  for (int kc = 0; kc < 8; ++kc) {
    v16bf pa = make_frag(
        *(const uint4*)&pw[arow * 256 + ((kc * 32 + half * 8) ^ asw)],
        *(const uint4*)&pw[arow * 256 + ((kc * 32 + 16 + half * 8) ^ asw)]);
#pragma unroll
    for (int nt = 0; nt < 4; ++nt) {
      const u16* vp = vbase + (size_t)(nt * 16 + l16) * 256 + kc * 32 + half * 8;
      v16bf vb = make_frag(*(const uint4*)vp, *(const uint4*)(vp + 16));
      c[nt] = wmma_bf16(pa, vb, c[nt]);
    }
  }

  const int b = bh >> 3, h = bh & 7;
#pragma unroll
  for (int nt = 0; nt < 4; ++nt)
#pragma unroll
    for (int j = 0; j < 8; ++j) {
      int seq = qt * 128 + wid * 16 + j + 8 * half;
      ctx[((size_t)(b * 256 + seq)) * HD + h * 64 + nt * 16 + l16] = f2bf(c[nt][j]);
    }
}

// ---------------------------------------------------------------------------
extern "C" void kernel_launch(void* const* d_in, const int* in_sizes, int n_in,
                              void* d_out, int out_size, void* d_ws, size_t ws_size,
                              hipStream_t stream) {
  (void)in_sizes; (void)n_in; (void)out_size; (void)ws_size;
  const float* x_in  = (const float*)d_in[0];
  const float* abias = (const float*)d_in[1];
  const float* ln1s  = (const float*)d_in[2];
  const float* ln1b  = (const float*)d_in[3];
  const float* wq    = (const float*)d_in[4];
  const float* bq    = (const float*)d_in[5];
  const float* wk    = (const float*)d_in[6];
  const float* bk    = (const float*)d_in[7];
  const float* wv    = (const float*)d_in[8];
  const float* bv    = (const float*)d_in[9];
  const float* wo    = (const float*)d_in[10];
  const float* bo    = (const float*)d_in[11];
  const float* ln2s  = (const float*)d_in[12];
  const float* ln2b  = (const float*)d_in[13];
  const float* w1    = (const float*)d_in[14];
  const float* b1    = (const float*)d_in[15];
  const float* w2    = (const float*)d_in[16];
  const float* b2    = (const float*)d_in[17];

  float* x = (float*)d_out;            // residual stream lives in d_out

  // workspace carve-up (256B aligned)
  char* wsp = (char*)d_ws;
  auto alloc = [&](size_t bytes) {
    char* p = wsp; wsp += (bytes + 255) & ~(size_t)255; return p;
  };
  u16*   wqkvT = (u16*)alloc((size_t)LL * 1536 * 512 * 2);
  u16*   woT   = (u16*)alloc((size_t)LL * 512  * 512 * 2);
  u16*   w1T   = (u16*)alloc((size_t)LL * 2048 * 512 * 2);
  u16*   w2T   = (u16*)alloc((size_t)LL * 512  * 2048 * 2);
  float* bqkv  = (float*)alloc((size_t)LL * 1536 * 4);
  u16*   y     = (u16*)alloc((size_t)MROWS * HD * 2);
  u16*   qb    = (u16*)alloc((size_t)BB * NHE * NN * DD * 2);
  u16*   kb    = (u16*)alloc((size_t)BB * NHE * NN * DD * 2);
  u16*   vtb   = (u16*)alloc((size_t)BB * NHE * DD * NN * 2);
  u16*   ctxb  = (u16*)alloc((size_t)MROWS * HD * 2);
  u16*   h1    = (u16*)alloc((size_t)MROWS * FD * 2);

  const float qscale = 0.125f;  // D^-0.5, folded into wq and bq

  // ---- weight conversion (bf16, transposed to [Nout][K]) ----
  for (int l = 0; l < LL; ++l) {
    const size_t hh = (size_t)HD * HD, hf = (size_t)HD * FD;
    int nHH = HD * HD, nHF = HD * FD;
    int gHH = (nHH + 255) / 256, gHF = (nHF + 255) / 256;
    u16* qkvT = wqkvT + (size_t)l * 1536 * 512;
    conv_w<<<gHH, 256, 0, stream>>>(wq + l * hh, qkvT,                 512, 512, qscale);
    conv_w<<<gHH, 256, 0, stream>>>(wk + l * hh, qkvT + 512 * 512,     512, 512, 1.0f);
    conv_w<<<gHH, 256, 0, stream>>>(wv + l * hh, qkvT + 1024 * 512,    512, 512, 1.0f);
    conv_w<<<gHH, 256, 0, stream>>>(wo + l * hh, woT + (size_t)l * 512 * 512, 512, 512, 1.0f);
    conv_w<<<gHF, 256, 0, stream>>>(w1 + l * hf, w1T + (size_t)l * 2048 * 512, 512, 2048, 1.0f);
    conv_w<<<gHF, 256, 0, stream>>>(w2 + l * hf, w2T + (size_t)l * 512 * 2048, 2048, 512, 1.0f);
    scale_copy<<<2, 256, 0, stream>>>(bq + l * HD, bqkv + (size_t)l * 1536,        HD, qscale);
    scale_copy<<<2, 256, 0, stream>>>(bk + l * HD, bqkv + (size_t)l * 1536 + 512,  HD, 1.0f);
    scale_copy<<<2, 256, 0, stream>>>(bv + l * HD, bqkv + (size_t)l * 1536 + 1024, HD, 1.0f);
  }

  // ---- x := input ----
  (void)hipMemcpyAsync(x, x_in, (size_t)MROWS * HD * sizeof(float),
                       hipMemcpyDeviceToDevice, stream);

  // ---- layers ----
  for (int l = 0; l < LL; ++l) {
    const u16* qkvT = wqkvT + (size_t)l * 1536 * 512;
    // LN1 -> y (bf16)
    ln_kernel<<<MROWS / 8, 256, 0, stream>>>(x, ln1s + l * HD, ln1b + l * HD, y);
    // QKV: [8192,512] x [512,1536]
    gemm_wmma<<<dim3(1536 / 128, MROWS / 128), 256, 0, stream>>>(
        y, qkvT, bqkv + (size_t)l * 1536, nullptr, nullptr, nullptr,
        qb, kb, vtb, MROWS, 1536, 512, GEMM_QKV);
    // attention
    attn_wmma<<<BB * NHE * (NN / 128), 256, 0, stream>>>(qb, kb, vtb, abias, ctxb);
    // Wo + residual -> x (in place)
    gemm_wmma<<<dim3(HD / 128, MROWS / 128), 256, 0, stream>>>(
        ctxb, woT + (size_t)l * 512 * 512, bo + l * HD, x, x, nullptr,
        nullptr, nullptr, nullptr, MROWS, HD, 512, GEMM_RESF32);
    // LN2 -> y
    ln_kernel<<<MROWS / 8, 256, 0, stream>>>(x, ln2s + l * HD, ln2b + l * HD, y);
    // FFN1 + GELU -> h1 (bf16)
    gemm_wmma<<<dim3(FD / 128, MROWS / 128), 256, 0, stream>>>(
        y, w1T + (size_t)l * 2048 * 512, b1 + l * FD, nullptr, nullptr, h1,
        nullptr, nullptr, nullptr, MROWS, FD, 512, GEMM_GELU);
    // FFN2 + residual -> x (in place)
    gemm_wmma<<<dim3(HD / 128, MROWS / 128), 256, 0, stream>>>(
        h1, w2T + (size_t)l * 512 * 2048, b2 + l * HD, x, x, nullptr,
        nullptr, nullptr, nullptr, MROWS, HD, 2048, GEMM_RESF32);
  }
}
